// AlternativeSelfAttention_19078244728945
// MI455X (gfx1250) — matrix operand order repaired and checked by reading, not compiled, not run on hardware
//
#include <hip/hip_runtime.h>

// ---------------------------------------------------------------------------
// MHA forward on gfx1250 (CDNA5), all matmuls via v_wmma_f32_16x16x32_bf16.
//   N=4 batch, S=1024 seq, E=1024 embed, H=16 heads, D=64 head_dim
// Stage 1: per-head projections q,k,v  (V stored transposed per (n,h))
// Stage 2: flash-style attention, 64-key blocks (16 WMMA per softmax update)
// Stage 3: output GEMM  out = AO @ Wu^T + bu   (fp32 out)
// ---------------------------------------------------------------------------

#define DI static __device__ __forceinline__

typedef __attribute__((ext_vector_type(16))) __bf16 v16bf;
typedef __attribute__((ext_vector_type(8)))  __bf16 v8bf;
typedef __attribute__((ext_vector_type(8)))  float  v8f;

constexpr int N_ = 4, S_ = 1024, E_ = 1024, H_ = 16, D_ = 64;
constexpr float SCALE  = 0.03125f;   // 1/sqrt(E) = 1/32
constexpr float NEGBIG = -3.0e18f;   // ~ -1e20/32 (post-scale masked logit)

DI v8f wmma_bf16(v16bf a, v16bf b, v8f c) {
  // D = A(16x32 bf16) * B(32x16 bf16) + C(16x16 f32)
  return __builtin_amdgcn_wmma_f32_16x16x32_bf16(false, a, false, b,
                                                 (short)0, c, false, false);
}

DI v8f zero8() {
  v8f z;
#pragma unroll
  for (int j = 0; j < 8; ++j) z[j] = 0.0f;
  return z;
}

DI v16bf join16(v8bf lo, v8bf hi) {
  v16bf r;
#pragma unroll
  for (int j = 0; j < 8; ++j) { r[j] = lo[j]; r[8 + j] = hi[j]; }
  return r;
}

// A-matrix 16x32 bf16 fragment. Lane: m = lane&15, half = lane>>4.
// Element e: K = half*16 + e  (16 contiguous values).
DI v16bf afrag_from_f32(const float* __restrict__ row, int ks, int half) {
  v16bf a;
#pragma unroll
  for (int g = 0; g < 2; ++g) {
    const int kb = ks * 32 + half * 16 + g * 8;
#pragma unroll
    for (int j = 0; j < 8; ++j) a[g * 8 + j] = (__bf16)row[kb + j];
  }
  return a;
}

DI v16bf afrag_from_bf16(const __bf16* __restrict__ row, int ks, int half) {
  v8bf lo = *(const v8bf*)(row + ks * 32 + half * 16);
  v8bf hi = *(const v8bf*)(row + ks * 32 + half * 16 + 8);
  return join16(lo, hi);
}

// B-matrix 32x16 bf16 fragment. Lane: n = lane&15 (fixed column), half=lane>>4.
// Element e: K = half*16 + e  -> 16 contiguous values at *p.
DI v16bf bfrag_from_bf16(const __bf16* __restrict__ p) {
  v8bf lo = *(const v8bf*)p;
  v8bf hi = *(const v8bf*)(p + 8);
  return join16(lo, hi);
}

DI v16bf bfrag_from_f32(const float* __restrict__ p) {
  v16bf b;
#pragma unroll
  for (int j = 0; j < 16; ++j) b[j] = (__bf16)p[j];
  return b;
}

// ---------------------------------------------------------------------------
// Stage 1: projections.  grid = N*H*3*(S/128), block = 256 (8 waves).
// Each wave computes a 16x64 tile:  X(16x64) @ W^T(64x64).
// proj 0 -> qp[n][h][s][d], 1 -> kp[n][h][s][d], 2 -> vpt[n][h][d][s]
// ---------------------------------------------------------------------------
__global__ __launch_bounds__(256) void proj_kernel(
    const float* __restrict__ q_in, const float* __restrict__ k_in,
    const float* __restrict__ v_in,
    const float* __restrict__ Wq, const float* __restrict__ Wk,
    const float* __restrict__ Wv,
    __bf16* __restrict__ qp, __bf16* __restrict__ kp,
    __bf16* __restrict__ vpt) {
  int idx = blockIdx.x;
  const int sb   = idx & 7;  idx >>= 3;
  const int proj = idx % 3;  idx /= 3;
  const int h    = idx & 15;
  const int n    = idx >> 4;

  const int w = threadIdx.x >> 5, lane = threadIdx.x & 31;
  const int m = lane & 15, half = lane >> 4;
  const int s0 = sb * 128 + w * 16;
  const int nh = n * H_ + h;

  const float* X = (proj == 0) ? q_in : (proj == 1) ? k_in : v_in;
  const float* W = (proj == 0) ? Wq   : (proj == 1) ? Wk   : Wv;

  const float* xrow = X + ((size_t)n * S_ + s0 + m) * E_ + h * D_;
  const v16bf a0 = afrag_from_f32(xrow, 0, half);
  const v16bf a1 = afrag_from_f32(xrow, 1, half);

#pragma unroll
  for (int nt = 0; nt < 4; ++nt) {
    const int nc = nt * 16 + m;                 // output feature (column)
    // B[k][nc] = W[nc][k]  (x @ W^T) -> 16 contiguous k per lane
    const v16bf b0 = bfrag_from_f32(W + nc * D_ + 0 * 32 + half * 16);
    const v16bf b1 = bfrag_from_f32(W + nc * D_ + 1 * 32 + half * 16);
    v8f c = zero8();
    c = wmma_bf16(a0, b0, c);
    c = wmma_bf16(a1, b1, c);

    if (proj == 2) {
      // transposed store: vpt[nh][d][s]; rows r are consecutive s -> 16B pack
      v8bf vs;
#pragma unroll
      for (int r = 0; r < 8; ++r) vs[r] = (__bf16)c[r];
      const size_t off = ((size_t)nh * D_ + nc) * S_ + s0 + half * 8;
      *(v8bf*)(vpt + off) = vs;
    } else {
      __bf16* outp = ((proj == 0) ? qp : kp) +
                     ((size_t)nh * S_ + s0) * D_ + nc;
#pragma unroll
      for (int r = 0; r < 8; ++r)
        outp[(size_t)(r + 8 * half) * D_] = (__bf16)c[r];
    }
  }
}

// ---------------------------------------------------------------------------
// Stage 2: flash attention, 64-key blocks.
// grid = N*H*(S/128), block = 256 (8 waves); wave owns a 16-row query tile.
// ---------------------------------------------------------------------------
__global__ __launch_bounds__(256) void attn_kernel(
    const __bf16* __restrict__ qp, const __bf16* __restrict__ kp,
    const __bf16* __restrict__ vpt, const unsigned char* __restrict__ mask,
    __bf16* __restrict__ ao) {
  __shared__ __align__(16) __bf16 lK[64 * 64];      // [key][d]     8 KB
  __shared__ __align__(16) __bf16 lV[64 * 64];      // [d][key]     8 KB
  __shared__ __align__(16) __bf16 lP[8][16 * 64];   // per-wave P  16 KB

  const int qb = blockIdx.x & 7;
  const int nh = blockIdx.x >> 3;
  const int h = nh & 15, n = nh >> 4;
  const int w = threadIdx.x >> 5, lane = threadIdx.x & 31;
  const int m = lane & 15, half = lane >> 4;
  const int tid = threadIdx.x;
  const int s0 = qb * 128 + w * 16;

  // Q fragments for the whole key loop (K-dim = 64 = 2 wmma k-steps)
  const __bf16* qrow = qp + ((size_t)nh * S_ + s0 + m) * D_;
  const v16bf aq0 = afrag_from_bf16(qrow, 0, half);
  const v16bf aq1 = afrag_from_bf16(qrow, 1, half);

  float mrun[8], lrun[8];
  v8f o[4];
#pragma unroll
  for (int r = 0; r < 8; ++r) { mrun[r] = -3.0e38f; lrun[r] = 0.0f; }
#pragma unroll
  for (int nt = 0; nt < 4; ++nt) o[nt] = zero8();

  const size_t mbase = (size_t)n * S_ * S_;   // mask[n][0][q][key]

  for (int kb = 0; kb < S_ / 64; ++kb) {
    __syncthreads();
    // cooperative stage: K block (64x64) and V^T block (64x64), bf16
#pragma unroll
    for (int i = 0; i < 2; ++i) {
      const int off = tid * 16 + i * 8;      // 0..4095, 8 elems per store
      const int row = off >> 6, col = off & 63;
      *(v8bf*)&lK[off] =
          *(const v8bf*)(kp + ((size_t)nh * S_ + kb * 64 + row) * D_ + col);
      *(v8bf*)&lV[off] =
          *(const v8bf*)(vpt + ((size_t)nh * D_ + row) * S_ + kb * 64 + col);
    }
    if (kb < S_ / 64 - 1)   // hint next K block toward the caches
      __builtin_prefetch(kp + ((size_t)nh * S_ + (kb + 1) * 64) * D_, 0, 0);
    __syncthreads();

    // scores: four 16x16 tiles (keys t*16 .. t*16+15 of this 64-block)
    float p[4][8];
#pragma unroll
    for (int t = 0; t < 4; ++t) {
      v8f s = zero8();
      const __bf16* krow = &lK[(t * 16 + m) * 64 + half * 16];
      s = wmma_bf16(aq0, bfrag_from_bf16(krow), s);
      s = wmma_bf16(aq1, bfrag_from_bf16(krow + 32), s);
#pragma unroll
      for (int r = 0; r < 8; ++r) {
        const int qi  = s0 + r + 8 * half;
        const int key = kb * 64 + t * 16 + m;
        const float x = s[r] * SCALE;
        p[t][r] = mask[mbase + (size_t)qi * S_ + key] ? x : NEGBIG;
      }
    }

    // online softmax update (rows live in 16-lane halves: m = r + 8*half)
#pragma unroll
    for (int r = 0; r < 8; ++r) {
      float v = fmaxf(fmaxf(p[0][r], p[1][r]), fmaxf(p[2][r], p[3][r]));
      v = fmaxf(v, __shfl_xor(v, 1));
      v = fmaxf(v, __shfl_xor(v, 2));
      v = fmaxf(v, __shfl_xor(v, 4));
      v = fmaxf(v, __shfl_xor(v, 8));
      const float mnew  = fmaxf(mrun[r], v);
      const float alpha = __expf(mrun[r] - mnew);
      mrun[r] = mnew;
#pragma unroll
      for (int t = 0; t < 4; ++t) p[t][r] = __expf(p[t][r] - mnew);
      float rs = (p[0][r] + p[1][r]) + (p[2][r] + p[3][r]);
      rs += __shfl_xor(rs, 1);
      rs += __shfl_xor(rs, 2);
      rs += __shfl_xor(rs, 4);
      rs += __shfl_xor(rs, 8);
      lrun[r] = lrun[r] * alpha + rs;
#pragma unroll
      for (int nt = 0; nt < 4; ++nt) o[nt][r] *= alpha;
    }

    // C-layout -> A-layout reshuffle of P(16x64) through per-wave LDS
    __bf16* pw = lP[w];
#pragma unroll
    for (int t = 0; t < 4; ++t)
#pragma unroll
      for (int r = 0; r < 8; ++r)
        pw[(r + 8 * half) * 64 + t * 16 + m] = (__bf16)p[t][r];
    __asm__ volatile("s_wait_dscnt 0" ::: "memory");
    const v16bf ap0 = afrag_from_bf16(&pw[m * 64], 0, half);
    const v16bf ap1 = afrag_from_bf16(&pw[m * 64], 1, half);

    // O += P(16x64) @ V(64x64)
#pragma unroll
    for (int nt = 0; nt < 4; ++nt) {
      const __bf16* vrow = &lV[(nt * 16 + m) * 64 + half * 16];
      o[nt] = wmma_bf16(ap0, bfrag_from_bf16(vrow), o[nt]);
      o[nt] = wmma_bf16(ap1, bfrag_from_bf16(vrow + 32), o[nt]);
    }
  }

  // normalize and store attention output (bf16, [n][s][h*64+d])
#pragma unroll
  for (int r = 0; r < 8; ++r) {
    const float rl = (lrun[r] > 0.0f) ? 1.0f / lrun[r] : 0.0f;
    const int qi = s0 + r + 8 * half;
#pragma unroll
    for (int nt = 0; nt < 4; ++nt)
      ao[((size_t)n * S_ + qi) * E_ + h * D_ + nt * 16 + m] =
          (__bf16)(o[nt][r] * rl);
  }
}

// ---------------------------------------------------------------------------
// Stage 3: out = AO(4096x1024 bf16) @ Wu^T + bu  (fp32 out)
// grid = (N*S/128)*(E/64) = 512, block = 256 (8 waves).
// Wu k-tiles converted to bf16 in LDS, 8x reuse per block.
// ---------------------------------------------------------------------------
__global__ __launch_bounds__(256) void out_gemm_kernel(
    const __bf16* __restrict__ ao, const float* __restrict__ Wu,
    const float* __restrict__ bu, float* __restrict__ out) {
  __shared__ __align__(16) __bf16 lW[64 * 32];   // [ncol_local][k_local]

  const int cb = blockIdx.x & 15, rb = blockIdx.x >> 4;
  const int w = threadIdx.x >> 5, lane = threadIdx.x & 31;
  const int m = lane & 15, half = lane >> 4;
  const int tid = threadIdx.x;
  const int r0 = rb * 128 + w * 16, n0 = cb * 64;

  v8f acc[4];
#pragma unroll
  for (int nt = 0; nt < 4; ++nt) acc[nt] = zero8();

  const __bf16* arow = ao + (size_t)(r0 + m) * E_;

  for (int ks = 0; ks < 32; ++ks) {
    __syncthreads();
    {   // stage Wu[n0..n0+63][ks*32..+31] as bf16
      const int ncl = tid >> 2, kl = (tid & 3) * 8;
      const float* src = Wu + (size_t)(n0 + ncl) * E_ + ks * 32 + kl;
      v8bf t;
#pragma unroll
      for (int j = 0; j < 8; ++j) t[j] = (__bf16)src[j];
      *(v8bf*)&lW[tid * 8] = t;
    }
    __syncthreads();

    const v16bf a = afrag_from_bf16(arow, ks, half);
#pragma unroll
    for (int nt = 0; nt < 4; ++nt) {
      const v16bf b = bfrag_from_bf16(&lW[(nt * 16 + m) * 32 + half * 16]);
      acc[nt] = wmma_bf16(a, b, acc[nt]);
    }
  }

#pragma unroll
  for (int nt = 0; nt < 4; ++nt) {
    const int col = n0 + nt * 16 + m;
    const float bias = bu[col];
#pragma unroll
    for (int r = 0; r < 8; ++r)
      out[(size_t)(r0 + r + 8 * half) * E_ + col] = acc[nt][r] + bias;
  }
}

// ---------------------------------------------------------------------------
extern "C" void kernel_launch(void* const* d_in, const int* in_sizes, int n_in,
                              void* d_out, int out_size, void* d_ws,
                              size_t ws_size, hipStream_t stream) {
  (void)in_sizes; (void)n_in; (void)out_size; (void)ws_size;

  const float* values = (const float*)d_in[0];
  const float* keys   = (const float*)d_in[1];
  const float* query  = (const float*)d_in[2];
  const unsigned char* mask = (const unsigned char*)d_in[3];
  const float* Wk = (const float*)d_in[4];
  const float* Wq = (const float*)d_in[5];
  const float* Wv = (const float*)d_in[6];
  const float* Wu = (const float*)d_in[7];
  const float* bu = (const float*)d_in[8];
  float* out = (float*)d_out;

  char* ws = (char*)d_ws;
  const size_t seg = (size_t)N_ * H_ * S_ * D_ * sizeof(__bf16);  // 8 MB
  __bf16* qp  = (__bf16*)(ws);
  __bf16* kp  = (__bf16*)(ws + seg);
  __bf16* vpt = (__bf16*)(ws + 2 * seg);
  __bf16* ao  = (__bf16*)(ws + 3 * seg);

  proj_kernel<<<dim3(N_ * H_ * 3 * (S_ / 128)), 256, 0, stream>>>(
      query, keys, values, Wq, Wk, Wv, qp, kp, vpt);
  attn_kernel<<<dim3(N_ * H_ * (S_ / 128)), 256, 0, stream>>>(
      qp, kp, vpt, mask, ao);
  out_gemm_kernel<<<dim3((N_ * S_ / 128) * (E_ / 64)), 256, 0, stream>>>(
      ao, Wu, bu, out);
}
